// MakeFusionLoss_68659347194380
// MI455X (gfx1250) — matrix operand order repaired and checked by reading, not compiled, not run on hardware
//
#include <hip/hip_runtime.h>
#include <hip/hip_bf16.h>

typedef __attribute__((ext_vector_type(16))) _Float16 v16h;
typedef __attribute__((ext_vector_type(8)))  float    v8f;

#define IMW 512
#define IMH 512
#define PLANE (IMW * IMH)
#define N3 (16 * 3 * PLANE)
#define C1F 0.0001f
#define C2F 0.0009f

// Zero-padded ('SAME') load
__device__ __forceinline__ float ldz(const float* p, int r, int c) {
    return (r >= 0 && r < IMH && c >= 0 && c < IMW) ? p[r * IMW + c] : 0.f;
}

// acc layout (doubles): 0=con 1=gx 2=gy 3=cb 4=cr 5=ssim_rgb 6=ssim_ir
__global__ void zero_acc_kernel(double* acc) {
    if (threadIdx.x < 8) acc[threadIdx.x] = 0.0;
}

// ---------------------------------------------------------------------------
// Kernel 1: con_loss + gradient_loss (3x3 sobel) + color_loss, streaming.
// ---------------------------------------------------------------------------
__global__ void __launch_bounds__(256) losses_kernel(const float* __restrict__ vis,
                                                     const float* __restrict__ ir,
                                                     const float* __restrict__ fuse,
                                                     double* __restrict__ acc) {
    int idx = blockIdx.x * 256 + threadIdx.x;
    float con = 0.f, agx = 0.f, agy = 0.f, acb = 0.f, acr = 0.f;
    if (idx < N3) {
        int x = idx & 511;
        int y = (idx >> 9) & 511;
        int p = idx >> 18;            // plane index b*3+c in [0,48)
        int c = p % 3;
        const float* V = vis  + (size_t)p * PLANE;
        const float* I = ir   + (size_t)p * PLANE;
        const float* F = fuse + (size_t)p * PLANE;

        float nv[3][3], ni[3][3], nf[3][3];
#pragma unroll
        for (int dy = 0; dy < 3; ++dy)
#pragma unroll
            for (int dx = 0; dx < 3; ++dx) {
                int r = y + dy - 1, cc = x + dx - 1;
                nv[dy][dx] = ldz(V, r, cc);
                ni[dy][dx] = ldz(I, r, cc);
                nf[dy][dx] = ldz(F, r, cc);
            }
        // XLA conv = cross-correlation (no flip)
        auto sx = [](const float (&n)[3][3]) {
            return (n[0][2] - n[0][0]) + 2.f * (n[1][2] - n[1][0]) + (n[2][2] - n[2][0]);
        };
        auto sy = [](const float (&n)[3][3]) {
            return (n[0][0] + 2.f * n[0][1] + n[0][2]) - (n[2][0] + 2.f * n[2][1] + n[2][2]);
        };
        float fgx = sx(nf), vgx = sx(nv), igx = sx(ni);
        float fgy = sy(nf), vgy = sy(nv), igy = sy(ni);
        agx = fabsf(fgx - fmaxf(vgx, igx));
        agy = fabsf(fgy - fmaxf(vgy, igy));
        con = fabsf(nf[1][1] - fmaxf(nv[1][1], ni[1][1]));

        if (c == 0) {  // color loss computed once per pixel (R plane threads)
            int o = y * IMW + x;
            float Rv = nv[1][1], Gv = V[PLANE + o], Bv = V[2 * PLANE + o];
            float Rf = nf[1][1], Gf = F[PLANE + o], Bf = F[2 * PLANE + o];
            float Yv = 0.299f * Rv + 0.587f * Gv + 0.114f * Bv;
            float Yf = 0.299f * Rf + 0.587f * Gf + 0.114f * Bf;
            // |(a-Y)*k + 0.5 - ((b-Y')*k + 0.5)| -> 0.5 cancels
            acr = fabsf((Rf - Yf) * 0.713f - (Rv - Yv) * 0.713f);
            acb = fabsf((Bf - Yf) * 0.564f - (Bv - Yv) * 0.564f);
        }
    }

    __shared__ float sred[5];
    if (threadIdx.x < 5) sred[threadIdx.x] = 0.f;
    __syncthreads();
    float vals[5] = {con, agx, agy, acb, acr};
#pragma unroll
    for (int i = 0; i < 5; ++i) {
        float v = vals[i];
#pragma unroll
        for (int o = 16; o > 0; o >>= 1) v += __shfl_xor(v, o, 32);
        if ((threadIdx.x & 31) == 0) atomicAdd(&sred[i], v);
    }
    __syncthreads();
    if (threadIdx.x < 5) atomicAdd(&acc[threadIdx.x], (double)sred[threadIdx.x]);
}

// ---------------------------------------------------------------------------
// Kernel 2: SSIM via WMMA. One wave = one 16x16 output tile of one plane pair.
// Separable 11-tap Gaussian: horizontal pass (2 WMMAs), vertical pass on the
// transpose (1 WMMA) -- D-matrix f32 layout of the horizontal pass matches the
// f16 A-matrix layout of H^T exactly, so only f32->f16 packing is needed.
// ---------------------------------------------------------------------------
__global__ void __launch_bounds__(32) ssim_kernel(const float* __restrict__ vis,
                                                  const float* __restrict__ ir,
                                                  const float* __restrict__ fuse,
                                                  double* __restrict__ acc) {
    const int lane  = threadIdx.x;         // 0..31, wave32
    const int bid   = blockIdx.x;          // 0..65535
    const int plane = bid >> 10;           // 0..63 (48 rgb pairs + 16 ir pairs)
    const int t     = bid & 1023;
    const int rBase = (t >> 5) << 4;
    const int cBase = (t & 31) << 4;

    const float *img1, *i2r, *i2g = nullptr, *i2b = nullptr;
    bool gray;
    if (plane < 48) {
        int b = plane / 3, c = plane % 3;
        img1 = vis  + (size_t)(b * 3 + c) * PLANE;
        i2r  = fuse + (size_t)(b * 3 + c) * PLANE;
        gray = false;
    } else {
        int b = plane - 48;
        img1 = ir   + (size_t)(b * 3) * PLANE;   // ir channel 0
        i2r  = fuse + (size_t)(b * 3) * PLANE;
        i2g  = i2r + PLANE;
        i2b  = i2g + PLANE;
        gray = true;
    }

    // 11-tap Gaussian (sigma=1.5), normalized
    float gw[11];
    {
        float s = 0.f;
#pragma unroll
        for (int i = 0; i < 11; ++i) {
            float d = (float)(i - 5);
            gw[i] = __expf(-d * d * (1.0f / 4.5f));
            s += gw[i];
        }
        float inv = 1.f / s;
#pragma unroll
        for (int i = 0; i < 11; ++i) gw[i] *= inv;
    }

    const int m    = lane & 15;
    const int half = lane >> 4;

    // Constant banded-Gaussian B operand: B[k,n] = g[k-n], 0 outside [0,10].
    // f16 B 32x16 layout: lane -> col n=lane&15; dword v -> K = 8*half + 2v (+8 if v>=4) (+1).
    v16h bmat;
#pragma unroll
    for (int v = 0; v < 8; ++v) {
        int K0 = 8 * half + 2 * v + (v >= 4 ? 8 : 0);
        int d0 = K0 - m, d1 = K0 + 1 - m;
        bmat[2 * v]     = (d0 >= 0 && d0 <= 10) ? (_Float16)gw[d0] : (_Float16)0.f;
        bmat[2 * v + 1] = (d1 >= 0 && d1 <= 10) ? (_Float16)gw[d1] : (_Float16)0.f;
    }

    // Load input region rows rBase-5 .. rBase+26, cols cBase-5 .. cBase+26.
    // A-matrix f16 16x32 layout: lane -> row M=lane&15; elements 0..7 = cols
    // base-5+8*half+0..7 ; elements 8..15 = +16.  Two row groups (16 rows each).
    float x1[2][16], x2[2][16];
#pragma unroll
    for (int rg = 0; rg < 2; ++rg) {
        int row = rBase - 5 + 16 * rg + m;
#pragma unroll
        for (int ch = 0; ch < 2; ++ch) {
            int c0 = cBase - 5 + 8 * half + 16 * ch;
#pragma unroll
            for (int e = 0; e < 8; ++e) {
                int col = c0 + e, j = ch * 8 + e;
                x1[rg][j] = ldz(img1, row, col);
                if (gray) {
                    x2[rg][j] = 0.2989f * ldz(i2r, row, col)
                              + 0.587f  * ldz(i2g, row, col)
                              + 0.114f  * ldz(i2b, row, col);
                } else {
                    x2[rg][j] = ldz(i2r, row, col);
                }
            }
        }
    }

    const v8f zero = {0.f, 0.f, 0.f, 0.f, 0.f, 0.f, 0.f, 0.f};
    v8f out[5];
#pragma unroll
    for (int q = 0; q < 5; ++q) {
        v8f h[2];
#pragma unroll
        for (int rg = 0; rg < 2; ++rg) {
            v16h a;
#pragma unroll
            for (int j = 0; j < 16; ++j) {
                float xa = x1[rg][j], xb = x2[rg][j];
                float val = (q == 0) ? xa
                          : (q == 1) ? xb
                          : (q == 2) ? xa * xa
                          : (q == 3) ? xb * xb
                                     : xa * xb;
                a[j] = (_Float16)val;
            }
            h[rg] = __builtin_amdgcn_wmma_f32_16x16x32_f16(
                false, a, false, bmat, (short)0, zero, false, false);
        }
        // Pack H^T directly from the two D matrices (layouts line up).
        v16h av;
#pragma unroll
        for (int j = 0; j < 8; ++j) {
            av[j]     = (_Float16)h[0][j];
            av[8 + j] = (_Float16)h[1][j];
        }
        out[q] = __builtin_amdgcn_wmma_f32_16x16x32_f16(
            false, av, false, bmat, (short)0, zero, false, false);
    }

    // SSIM map (transposed tile layout is irrelevant: we only need the mean).
    float s = 0.f;
#pragma unroll
    for (int e = 0; e < 8; ++e) {
        float mu1 = out[0][e], mu2 = out[1][e];
        float mu1sq = mu1 * mu1, mu2sq = mu2 * mu2, mu12 = mu1 * mu2;
        float s1  = out[2][e] - mu1sq;
        float s2  = out[3][e] - mu2sq;
        float s12 = out[4][e] - mu12;
        float num = (2.f * mu12 + C1F) * (2.f * s12 + C2F);
        float den = (mu1sq + mu2sq + C1F) * (s1 + s2 + C2F);
        s += num / den;
    }
#pragma unroll
    for (int o = 16; o > 0; o >>= 1) s += __shfl_xor(s, o, 32);
    if (lane == 0) atomicAdd(&acc[gray ? 6 : 5], (double)s);
}

// ---------------------------------------------------------------------------
// Kernel 3: combine partial sums into the scalar loss.
// ---------------------------------------------------------------------------
__global__ void finalize_kernel(const double* __restrict__ acc, float* __restrict__ out) {
    if (threadIdx.x == 0 && blockIdx.x == 0) {
        const double n3 = (double)N3;
        const double n1 = 16.0 * PLANE;
        double con_loss  = acc[0] / n3;
        double grad_loss = 0.5 * (acc[1] / n3) + 0.5 * (acc[2] / n3);
        double color     = acc[3] / n1 + acc[4] / n1;
        double ssim_rgb  = acc[5] / n3;
        double ssim_ir   = acc[6] / n1;
        double ssim_loss = 1.0 - (ssim_rgb + ssim_ir) * 0.5;
        out[0] = (float)(0.5 * con_loss + 0.2 * grad_loss + 1.0 * color + 0.1 * ssim_loss);
    }
}

extern "C" void kernel_launch(void* const* d_in, const int* in_sizes, int n_in,
                              void* d_out, int out_size, void* d_ws, size_t ws_size,
                              hipStream_t stream) {
    (void)in_sizes; (void)n_in; (void)out_size; (void)ws_size;
    const float* vis  = (const float*)d_in[0];
    const float* ir   = (const float*)d_in[1];
    const float* fuse = (const float*)d_in[2];
    double* acc = (double*)d_ws;
    float* out  = (float*)d_out;

    zero_acc_kernel<<<1, 32, 0, stream>>>(acc);
    losses_kernel<<<(N3 + 255) / 256, 256, 0, stream>>>(vis, ir, fuse, acc);
    ssim_kernel<<<64 * 1024, 32, 0, stream>>>(vis, ir, fuse, acc);
    finalize_kernel<<<1, 32, 0, stream>>>(acc, out);
}
